// Transformer_67095979098522
// MI455X (gfx1250) — matrix operand order repaired and checked by reading, not compiled, not run on hardware
//
#include <hip/hip_runtime.h>
#include <hip/hip_bf16.h>

// ---------------------------------------------------------------------------
// Transformer (6 enc + 6 dec layers, D=512, H=8, F=2048, B=8, S=T=512)
// CDNA5 / gfx1250: all GEMMs via v_wmma_f32_16x16x32_bf16 (wave32 WMMA),
// tiles staged with global_load_async_to_lds_b128 + double-buffered LDS.
// All B operands are pre-transposed to [N,K] so staging is pure b128 copies.
// ---------------------------------------------------------------------------

typedef __attribute__((ext_vector_type(16))) __bf16 v16bf;
typedef __attribute__((ext_vector_type(8)))  float  v8f;

#define Lc  6
#define Dc  512
#define Hc  8
#define DKc 64
#define Fc  2048
#define Bc  8
#define Sc  512
#define Tc  512
#define BSc (Bc * Sc)   // 4096 tokens

// GEMM tiling
#define BM 128
#define BK 64           // two WMMA k-steps per staged tile
#define LDSK (BK + 8)   // 72 halves = 144B row stride (16B aligned)

// flags
#define GF_BIAS     2
#define GF_RELU     4
#define GF_OUT_F32  8
#define GF_OUT_BF16 16

struct GemmP {
    const __bf16* A;      // [M,K] row-major (lda)
    const __bf16* B;      // [N,K] row-major (ldb)  -- i.e. B^T of the math B
    const float*  bias;
    float*        C32;
    __bf16*       C16;
    int M, N, K, lda, ldb, ldc, flags, Hdiv;
    long long sAb, sAh, sBb, sBh, sCb, sCh;  // per-(batch,head) element strides
};

// ---- CDNA5 async global->LDS staging -------------------------------------
__device__ __forceinline__ void async_ld_b128(void* lds, const void* gptr)
{
    // LDS aperture flat address: addr[31:0] is the LDS byte offset.
    unsigned l = (unsigned)(unsigned long long)lds;
    asm volatile("global_load_async_to_lds_b128 %0, %1, off"
                 :: "v"(l), "v"(gptr) : "memory");
}
__device__ __forceinline__ void wait_async0()
{
    asm volatile("s_wait_asynccnt 0x0" ::: "memory");
}

// ---------------------------------------------------------------------------
// Tiled bf16 WMMA GEMM (fully aligned fast path: M%128==0, K%64==0,
// N % (32*NTILES) == 0).  256 threads = 8 waves arranged 4(m) x 2(n);
// each wave owns 32 x (16*NTILES) = 2 x NTILES tiles of 16x16 f32 acc.
// ---------------------------------------------------------------------------
template <int NTILES>
__global__ __launch_bounds__(256, 1) void gemm_bf16_wmma(GemmP p)
{
    constexpr int BN = 32 * NTILES;
    __shared__ __bf16 Ash[2][BM][LDSK];
    __shared__ __bf16 Bsh[2][BN][LDSK];

    const int tid   = threadIdx.x;
    const int lane  = tid & 31;
    const int wid   = tid >> 5;
    const int wm    = wid >> 1;       // 0..3  -> 32-row slab
    const int wn    = wid & 1;        // 0..1  -> (16*NTILES)-col slab
    const int lanem = lane & 15;
    const int laneh = lane >> 4;      // 0 / 1

    const int z  = blockIdx.z;
    const int zb = z / p.Hdiv;
    const int zh = z % p.Hdiv;
    const __bf16* Ag = p.A + (long long)zb * p.sAb + (long long)zh * p.sAh;
    const __bf16* Bg = p.B + (long long)zb * p.sBb + (long long)zh * p.sBh;
    const long long coff = (long long)zb * p.sCb + (long long)zh * p.sCh;

    const int row0 = blockIdx.y * BM;
    const int col0 = blockIdx.x * BN;

    v8f acc[2][NTILES];
#pragma unroll
    for (int mt = 0; mt < 2; ++mt)
#pragma unroll
        for (int nt = 0; nt < NTILES; ++nt)
            acc[mt][nt] = (v8f){0.f, 0.f, 0.f, 0.f, 0.f, 0.f, 0.f, 0.f};

    // stage tile t into LDS buffer `buf` (async, per-wave ASYNCcnt).
    // Fixed trip counts -> unconditional async issues (no exec masking).
    auto stage = [&](int t, int buf) {
        const int kk = t * BK;
#pragma unroll
        for (int it = 0; it < (BM * (BK / 8)) / 256; ++it) {   // 16B segments
            const int s = tid + it * 256;
            const int r = s >> 3, sg = (s & 7) * 8;
            async_ld_b128(&Ash[buf][r][sg],
                          Ag + (long long)(row0 + r) * p.lda + kk + sg);
        }
#pragma unroll
        for (int it = 0; it < (BN * (BK / 8)) / 256; ++it) {
            const int s = tid + it * 256;
            const int n = s >> 3, sg = (s & 7) * 8;
            async_ld_b128(&Bsh[buf][n][sg],
                          Bg + (long long)(col0 + n) * p.ldb + kk + sg);
        }
    };

    // two WMMA k-steps from LDS buffer `buf`
    auto compute = [&](int buf) {
#pragma unroll
        for (int kh = 0; kh < 2; ++kh) {
            const int kc = kh * 32;
            v16bf af[2];
#pragma unroll
            for (int mt = 0; mt < 2; ++mt) {
                const __bf16* ap =
                    &Ash[buf][wm * 32 + mt * 16 + lanem][kc + laneh * 8];
                union { v16bf v; uint4 q[2]; } ua;
                ua.q[0] = *(const uint4*)(ap);        // K = base .. base+7
                ua.q[1] = *(const uint4*)(ap + 16);   // K = base+16 .. base+23
                af[mt] = ua.v;
            }
#pragma unroll
            for (int nt = 0; nt < NTILES; ++nt) {
                const __bf16* bp =
                    &Bsh[buf][wn * (16 * NTILES) + nt * 16 + lanem][kc + laneh * 8];
                union { v16bf v; uint4 q[2]; } ub;
                ub.q[0] = *(const uint4*)(bp);        // K = base .. base+7
                ub.q[1] = *(const uint4*)(bp + 16);   // K = base+16 .. base+23
#pragma unroll
                for (int mt = 0; mt < 2; ++mt)
                    acc[mt][nt] = __builtin_amdgcn_wmma_f32_16x16x32_bf16(
                        false, af[mt], false, ub.v, (short)0, acc[mt][nt],
                        false, false);
            }
        }
    };

    const int nk = p.K / BK;
    stage(0, 0);
    wait_async0();
    __syncthreads();
    for (int t = 0; t < nk; ++t) {
        if (t + 1 < nk) stage(t + 1, (t + 1) & 1);
        compute(t & 1);
        wait_async0();          // next tile fully in LDS
        __syncthreads();        // all waves done reading current buffer
    }

    // epilogue: lane 0-15: N=lane, M=i ; lane 16-31: N=lane-16, M=i+8
#pragma unroll
    for (int mt = 0; mt < 2; ++mt) {
#pragma unroll
        for (int nt = 0; nt < NTILES; ++nt) {
            const int gcol = col0 + wn * (16 * NTILES) + nt * 16 + lanem;
            const float bv = (p.flags & GF_BIAS) ? p.bias[gcol] : 0.f;
#pragma unroll
            for (int i = 0; i < 8; ++i) {
                const int grow = row0 + wm * 32 + mt * 16 + laneh * 8 + i;
                float v = acc[mt][nt][i] + bv;
                if (p.flags & GF_RELU) v = fmaxf(v, 0.f);
                const long long off = coff + (long long)grow * p.ldc + gcol;
                if (p.flags & GF_OUT_F32)  p.C32[off] = v;
                if (p.flags & GF_OUT_BF16) p.C16[off] = (__bf16)v;
            }
        }
    }
}

// ---------------------------------------------------------------------------
// softmax over rows of scores, fused scale + optional additive mask.
// ---------------------------------------------------------------------------
__global__ __launch_bounds__(256) void softmax_kernel(
    const float* __restrict__ S, const float* __restrict__ mask,
    __bf16* __restrict__ P, int cols, int qmod, float scale)
{
    __shared__ float red[256];
    const long long row = blockIdx.x;
    const int tid = threadIdx.x;
    const float* sr = S + row * cols;
    const float* mr = mask ? (mask + (long long)(row % qmod) * cols) : nullptr;

    float m = -3.0e38f;
    for (int c = tid; c < cols; c += 256) {
        float v = sr[c] * scale + (mr ? mr[c] : 0.f);
        m = fmaxf(m, v);
    }
    red[tid] = m; __syncthreads();
    for (int o = 128; o > 0; o >>= 1) { if (tid < o) red[tid] = fmaxf(red[tid], red[tid + o]); __syncthreads(); }
    m = red[0]; __syncthreads();

    float s = 0.f;
    for (int c = tid; c < cols; c += 256) {
        float v = sr[c] * scale + (mr ? mr[c] : 0.f);
        s += __expf(v - m);
    }
    red[tid] = s; __syncthreads();
    for (int o = 128; o > 0; o >>= 1) { if (tid < o) red[tid] += red[tid + o]; __syncthreads(); }
    const float inv = 1.f / red[0];

    for (int c = tid; c < cols; c += 256) {
        float v = sr[c] * scale + (mr ? mr[c] : 0.f);
        P[row * cols + c] = (__bf16)(__expf(v - m) * inv);
    }
}

// ---------------------------------------------------------------------------
// y = a * (x + r - mean) / (std_unbiased + 1e-6) + b ; writes f32 + bf16.
// ---------------------------------------------------------------------------
__global__ __launch_bounds__(256) void add_ln_kernel(
    const float* __restrict__ X, const float* __restrict__ R,
    const float* __restrict__ ga, const float* __restrict__ gb,
    float* __restrict__ Y32, __bf16* __restrict__ Y16, int Dd)
{
    __shared__ float red[256];
    const long long row = blockIdx.x;
    const int tid = threadIdx.x;
    const float* xr = X + row * Dd;
    const float* rr = R + row * Dd;

    float hv[8];
    int cnt = 0; float s = 0.f;
    for (int c = tid; c < Dd; c += 256) { float h = xr[c] + rr[c]; hv[cnt++] = h; s += h; }
    red[tid] = s; __syncthreads();
    for (int o = 128; o > 0; o >>= 1) { if (tid < o) red[tid] += red[tid + o]; __syncthreads(); }
    const float mean = red[0] / (float)Dd; __syncthreads();

    s = 0.f; cnt = 0;
    for (int c = tid; c < Dd; c += 256) { float d = hv[cnt++] - mean; s += d * d; }
    red[tid] = s; __syncthreads();
    for (int o = 128; o > 0; o >>= 1) { if (tid < o) red[tid] += red[tid + o]; __syncthreads(); }
    const float var = red[0] / (float)(Dd - 1);
    const float inv = 1.f / (sqrtf(var) + 1e-6f);

    cnt = 0;
    for (int c = tid; c < Dd; c += 256) {
        float y = ga[c] * (hv[cnt++] - mean) * inv + gb[c];
        Y32[row * Dd + c] = y;
        Y16[row * Dd + c] = (__bf16)y;
    }
}

// ---------------------------------------------------------------------------
// weight pack / convert kernels (all emit TRANSPOSED bf16: [N,K] row-major)
// ---------------------------------------------------------------------------
__global__ __launch_bounds__(256) void pack_hdk_t_kernel(
    const float* __restrict__ W, __bf16* __restrict__ out, long long total)
{
    // W[h][d][k] -> out[(h*DK+k)][d]   (B^T for the projection GEMM)
    long long i = (long long)blockIdx.x * 256 + threadIdx.x;
    if (i >= total) return;
    const int k = (int)(i % DKc);
    const long long t = i / DKc;
    const int d = (int)(t % Dc);
    const int h = (int)(t / Dc);
    out[((long long)h * DKc + k) * Dc + d] = (__bf16)W[i];
}

__global__ __launch_bounds__(256) void cvt_t_kernel(
    const float* __restrict__ in, __bf16* __restrict__ out, int R, int C)
{
    // in[R][C] -> out[C][R]
    long long i = (long long)blockIdx.x * 256 + threadIdx.x;
    if (i >= (long long)R * C) return;
    const int r = (int)(i / C);
    const int c = (int)(i % C);
    out[(long long)c * R + r] = (__bf16)in[i];
}

__global__ __launch_bounds__(256) void cvt_kernel(
    const float* __restrict__ in, __bf16* __restrict__ out, long long n)
{
    long long i = (long long)blockIdx.x * 256 + threadIdx.x;
    if (i < n) out[i] = (__bf16)in[i];
}

// per-batch transpose of v: [B][S][D] -> [B][D][S]
__global__ __launch_bounds__(256) void transpose_v_kernel(
    const __bf16* __restrict__ in, __bf16* __restrict__ out)
{
    long long i = (long long)blockIdx.x * 256 + threadIdx.x;
    if (i >= (long long)Bc * Sc * Dc) return;
    const int c = (int)(i % Dc);
    const long long t = i / Dc;
    const int s = (int)(t % Sc);
    const int b = (int)(t / Sc);
    out[(long long)b * Dc * Sc + (long long)c * Sc + s] = in[i];
}

// ---------------------------------------------------------------------------
// host orchestration
// ---------------------------------------------------------------------------
static char* ws_alloc(char*& p, size_t bytes)
{
    char* r = p;
    p += (bytes + 255) & ~(size_t)255;
    return r;
}

template <int NTILES>
static void gemm_launch(hipStream_t st, const __bf16* A, const __bf16* B, const float* bias,
                        float* C32, __bf16* C16, int M, int N, int K,
                        int lda, int ldb, int ldc,
                        int nz, int Hdiv,
                        long long sAb, long long sAh, long long sBb, long long sBh,
                        long long sCb, long long sCh, int flags)
{
    GemmP p;
    p.A = A; p.B = B; p.bias = bias; p.C32 = C32; p.C16 = C16;
    p.M = M; p.N = N; p.K = K; p.lda = lda; p.ldb = ldb; p.ldc = ldc;
    p.flags = flags; p.Hdiv = (Hdiv > 0 ? Hdiv : 1);
    p.sAb = sAb; p.sAh = sAh; p.sBb = sBb; p.sBh = sBh; p.sCb = sCb; p.sCh = sCh;
    const int BNv = 32 * NTILES;
    dim3 g((N + BNv - 1) / BNv, (M + BM - 1) / BM, nz);
    gemm_bf16_wmma<NTILES><<<g, dim3(256), 0, st>>>(p);
}

static void cvt_launch(hipStream_t st, const float* in, __bf16* out, long long n)
{
    cvt_kernel<<<(unsigned)((n + 255) / 256), 256, 0, st>>>(in, out, n);
}
static void cvt_t_launch(hipStream_t st, const float* in, __bf16* out, int R, int C)
{
    cvt_t_kernel<<<(unsigned)(((long long)R * C + 255) / 256), 256, 0, st>>>(in, out, R, C);
}

extern "C" void kernel_launch(void* const* d_in, const int* in_sizes, int n_in,
                              void* d_out, int out_size, void* d_ws, size_t ws_size,
                              hipStream_t stream)
{
    (void)in_sizes; (void)n_in; (void)out_size; (void)ws_size;

    // ---- inputs (setup_inputs order) ----
    const float* src      = (const float*)d_in[0];
    const float* tgt      = (const float*)d_in[1];
    const float* tgt_mask = (const float*)d_in[2];
    const float* enc_Wq   = (const float*)d_in[3];
    const float* enc_Wk   = (const float*)d_in[4];
    const float* enc_Wv   = (const float*)d_in[5];
    const float* enc_Wo   = (const float*)d_in[6];
    const float* enc_ln1a = (const float*)d_in[7];
    const float* enc_ln1b = (const float*)d_in[8];
    const float* enc_W1   = (const float*)d_in[9];
    const float* enc_b1   = (const float*)d_in[10];
    const float* enc_W2   = (const float*)d_in[11];
    const float* enc_b2   = (const float*)d_in[12];
    const float* enc_ln2a = (const float*)d_in[13];
    const float* enc_ln2b = (const float*)d_in[14];
    const float* dec_sWq  = (const float*)d_in[15];
    const float* dec_sWk  = (const float*)d_in[16];
    const float* dec_sWv  = (const float*)d_in[17];
    const float* dec_sWo  = (const float*)d_in[18];
    const float* dec_ln1a = (const float*)d_in[19];
    const float* dec_ln1b = (const float*)d_in[20];
    const float* dec_cWq  = (const float*)d_in[21];
    const float* dec_cWk  = (const float*)d_in[22];
    const float* dec_cWv  = (const float*)d_in[23];
    const float* dec_cWo  = (const float*)d_in[24];
    const float* dec_ln2a = (const float*)d_in[25];
    const float* dec_ln2b = (const float*)d_in[26];
    const float* dec_W1   = (const float*)d_in[27];
    const float* dec_b1   = (const float*)d_in[28];
    const float* dec_W2   = (const float*)d_in[29];
    const float* dec_b2   = (const float*)d_in[30];
    const float* dec_ln3a = (const float*)d_in[31];
    const float* dec_ln3b = (const float*)d_in[32];

    // ---- workspace carve-up ----
    char* w = (char*)d_ws;
    __bf16* Wq_t  = (__bf16*)ws_alloc(w, (size_t)Dc * Dc * 2);
    __bf16* Wk_t  = (__bf16*)ws_alloc(w, (size_t)Dc * Dc * 2);
    __bf16* Wv_t  = (__bf16*)ws_alloc(w, (size_t)Dc * Dc * 2);
    __bf16* Wo_t  = (__bf16*)ws_alloc(w, (size_t)Dc * Dc * 2);
    __bf16* W1_t  = (__bf16*)ws_alloc(w, (size_t)Dc * Fc * 2);
    __bf16* W2_t  = (__bf16*)ws_alloc(w, (size_t)Fc * Dc * 2);
    __bf16* X16   = (__bf16*)ws_alloc(w, (size_t)BSc * Dc * 2);
    float*  x32   = (float*) ws_alloc(w, (size_t)BSc * Dc * 4);
    __bf16* q16   = (__bf16*)ws_alloc(w, (size_t)BSc * Dc * 2);
    __bf16* k16   = (__bf16*)ws_alloc(w, (size_t)BSc * Dc * 2);
    __bf16* v16   = (__bf16*)ws_alloc(w, (size_t)BSc * Dc * 2);
    __bf16* vT16  = (__bf16*)ws_alloc(w, (size_t)BSc * Dc * 2);
    float*  sc32  = (float*) ws_alloc(w, (size_t)Bc * Hc * Sc * Sc * 4);
    __bf16* pr16  = (__bf16*)ws_alloc(w, (size_t)Bc * Hc * Sc * Sc * 2);
    __bf16* o16   = (__bf16*)ws_alloc(w, (size_t)BSc * Dc * 2);
    float*  sub32 = (float*) ws_alloc(w, (size_t)BSc * Dc * 4);
    __bf16* h16   = (__bf16*)ws_alloc(w, (size_t)BSc * Fc * 2);
    __bf16* mem16 = (__bf16*)ws_alloc(w, (size_t)BSc * Dc * 2);

    const long long WHDK = (long long)Hc * Dc * DKc;
    const unsigned  packGrid = (unsigned)((WHDK + 255) / 256);
    const float     scale = 0.125f;                      // 1/sqrt(64)
    const int       nzBH = Bc * Hc;
    const long long sBD = (long long)Sc * Dc;            // per-batch token stride
    const long long sSS = (long long)Sc * Sc;
    const long long sDS = (long long)Dc * Sc;

    auto attention = [&](const __bf16* qx16, const __bf16* kvx16,
                         const float* Wq, const float* Wk, const float* Wv, const float* Wo,
                         const float* mask) {
        pack_hdk_t_kernel<<<packGrid, 256, 0, stream>>>(Wq, Wq_t, WHDK);
        pack_hdk_t_kernel<<<packGrid, 256, 0, stream>>>(Wk, Wk_t, WHDK);
        pack_hdk_t_kernel<<<packGrid, 256, 0, stream>>>(Wv, Wv_t, WHDK);
        cvt_t_launch(stream, Wo, Wo_t, Dc, Dc);
        // q/k/v projections: [BS,D] x [D,D]^T-layout -> bf16 [B,S,H,DK]
        gemm_launch<4>(stream, qx16,  Wq_t, nullptr, nullptr, q16, BSc, Dc, Dc, Dc, Dc, Dc,
                       1, 1, 0, 0, 0, 0, 0, 0, GF_OUT_BF16);
        gemm_launch<4>(stream, kvx16, Wk_t, nullptr, nullptr, k16, BSc, Dc, Dc, Dc, Dc, Dc,
                       1, 1, 0, 0, 0, 0, 0, 0, GF_OUT_BF16);
        gemm_launch<4>(stream, kvx16, Wv_t, nullptr, nullptr, v16, BSc, Dc, Dc, Dc, Dc, Dc,
                       1, 1, 0, 0, 0, 0, 0, 0, GF_OUT_BF16);
        transpose_v_kernel<<<(unsigned)(((long long)BSc * Dc + 255) / 256), 256, 0, stream>>>(
            v16, vT16);
        // scores[b,h] = q[b,:,h,:] * k[b,:,h,:]^T  (k16 is already [N=sk][K=dk])
        gemm_launch<4>(stream, q16, k16, nullptr, sc32, nullptr, Sc, Sc, DKc, Dc, Dc, Sc,
                       nzBH, Hc, sBD, DKc, sBD, DKc, (long long)Hc * sSS, sSS,
                       GF_OUT_F32);
        softmax_kernel<<<(unsigned)(Bc * Hc * Sc), 256, 0, stream>>>(
            sc32, mask, pr16, Sc, Sc, scale);
        // o[b,:,h,:] = probs[b,h] * v[b,:,h,:]   (B^T = vT16[b][dk][sk]); N=64 -> BN=64
        gemm_launch<2>(stream, pr16, vT16, nullptr, nullptr, o16, Sc, DKc, Sc, Sc, Sc, Dc,
                       nzBH, Hc, (long long)Hc * sSS, sSS, sDS, (long long)DKc * Sc,
                       sBD, DKc, GF_OUT_BF16);
        // concat-heads @ Wo -> sub32 (f32 sublayer output)
        gemm_launch<4>(stream, o16, Wo_t, nullptr, sub32, nullptr, BSc, Dc, Dc, Dc, Dc, Dc,
                       1, 1, 0, 0, 0, 0, 0, 0, GF_OUT_F32);
    };

    auto ffn = [&](const float* W1, const float* b1, const float* W2, const float* b2) {
        cvt_t_launch(stream, W1, W1_t, Dc, Fc);   // -> [F,D]
        cvt_t_launch(stream, W2, W2_t, Fc, Dc);   // -> [D,F]
        gemm_launch<4>(stream, X16, W1_t, b1, nullptr, h16, BSc, Fc, Dc, Dc, Dc, Fc,
                       1, 1, 0, 0, 0, 0, 0, 0, GF_BIAS | GF_RELU | GF_OUT_BF16);
        gemm_launch<4>(stream, h16, W2_t, b2, sub32, nullptr, BSc, Dc, Fc, Fc, Fc, Dc,
                       1, 1, 0, 0, 0, 0, 0, 0, GF_BIAS | GF_OUT_F32);
    };

    auto layer_norm = [&](const float* res, const float* ga, const float* gb,
                          float* y32, __bf16* y16) {
        add_ln_kernel<<<(unsigned)BSc, 256, 0, stream>>>(res, sub32, ga, gb, y32, y16, Dc);
    };

    // ================= encoder =================
    cvt_launch(stream, src, X16, (long long)BSc * Dc);
    const float* res32 = src;
    for (int i = 0; i < Lc; ++i) {
        attention(X16, X16, enc_Wq + i * WHDK, enc_Wk + i * WHDK, enc_Wv + i * WHDK,
                  enc_Wo + (long long)i * Dc * Dc, nullptr);
        layer_norm(res32, enc_ln1a + i * Dc, enc_ln1b + i * Dc, x32, X16);
        res32 = x32;
        ffn(enc_W1 + (long long)i * Dc * Fc, enc_b1 + i * Fc,
            enc_W2 + (long long)i * Fc * Dc, enc_b2 + i * Dc);
        layer_norm(res32, enc_ln2a + i * Dc, enc_ln2b + i * Dc, x32, X16);
    }
    hipMemcpyAsync(mem16, X16, (size_t)BSc * Dc * 2, hipMemcpyDeviceToDevice, stream);

    // ================= decoder =================
    cvt_launch(stream, tgt, X16, (long long)BSc * Dc);
    res32 = tgt;
    for (int i = 0; i < Lc; ++i) {
        // masked self-attention
        attention(X16, X16, dec_sWq + i * WHDK, dec_sWk + i * WHDK, dec_sWv + i * WHDK,
                  dec_sWo + (long long)i * Dc * Dc, tgt_mask);
        layer_norm(res32, dec_ln1a + i * Dc, dec_ln1b + i * Dc, x32, X16);
        res32 = x32;
        // cross-attention (k/v from encoder memory)
        attention(X16, mem16, dec_cWq + i * WHDK, dec_cWk + i * WHDK, dec_cWv + i * WHDK,
                  dec_cWo + (long long)i * Dc * Dc, nullptr);
        layer_norm(res32, dec_ln2a + i * Dc, dec_ln2b + i * Dc, x32, X16);
        res32 = x32;
        // feed-forward
        ffn(dec_W1 + (long long)i * Dc * Fc, dec_b1 + i * Fc,
            dec_W2 + (long long)i * Fc * Dc, dec_b2 + i * Dc);
        float* outY = (i == Lc - 1) ? (float*)d_out : x32;
        layer_norm(res32, dec_ln3a + i * Dc, dec_ln3b + i * Dc, outY, X16);
        res32 = x32;
    }
}
